// MultiHeadAttention_37177236914546
// MI455X (gfx1250) — matrix operand order repaired and checked by reading, not compiled, not run on hardware
//
#include <hip/hip_runtime.h>
#include <hip/hip_bf16.h>

// ---- problem constants (B=2, S=2048, D=1024, H=16, Hd=64) ----
#define S_LEN    2048
#define D_MODEL  1024
#define HEADS    16
#define HEAD_DIM 64
#define BATCH    2
#define M_ROWS   (BATCH * S_LEN)   // 4096
#define QKV_N    (3 * D_MODEL)     // 3072

typedef __bf16 bf16_t;
typedef __attribute__((ext_vector_type(16))) __bf16 v16bf;
typedef __attribute__((ext_vector_type(8)))  __bf16 v8bf;
typedef __attribute__((ext_vector_type(8)))  float  v8f;
typedef __attribute__((ext_vector_type(4)))  int    v4i;

// ---- CDNA5 async global->LDS path (guarded; fallback = sync copy) ----
#if __has_builtin(__builtin_amdgcn_global_load_async_to_lds_b128)
#define HAVE_ASYNC 1
#else
#define HAVE_ASYNC 0
#endif

typedef __attribute__((address_space(1))) v4i* as1_v4i_p;
typedef __attribute__((address_space(3))) v4i* as3_v4i_p;

__device__ __forceinline__ void async_b128(bf16_t* lds_dst, const bf16_t* gsrc) {
#if HAVE_ASYNC
  __builtin_amdgcn_global_load_async_to_lds_b128((as1_v4i_p)gsrc, (as3_v4i_p)lds_dst,
                                                 /*imm offset*/0, /*cpol*/0);
#else
  *(v8bf*)lds_dst = *(const v8bf*)gsrc;
#endif
}

#if HAVE_ASYNC
#if __has_builtin(__builtin_amdgcn_s_wait_asynccnt)
#define ASYNC_WAIT(n) __builtin_amdgcn_s_wait_asynccnt(n)
#else
#define ASYNC_WAIT(n) asm volatile("s_wait_asynccnt %0" ::"i"(n) : "memory")
#endif
#else
#define ASYNC_WAIT(n) do {} while (0)
#endif

// fp32 -> bf16, round-to-nearest-even (bit trick)
__device__ __forceinline__ bf16_t f2bf(float x) {
  unsigned u = __float_as_uint(x);
  unsigned r = (u + 0x7FFFu + ((u >> 16) & 1u)) >> 16;
  unsigned short h = (unsigned short)r;
  return __builtin_bit_cast(bf16_t, h);
}

union AFragU { v16bf v; v8bf h[2]; };

// A fragment: 16x32 bf16 (MxK), source row-major [row][k], stride in elements.
// lanes 0-15 -> M=lane, K = {0..7, 16..23}; lanes 16-31 -> M=lane-16, K = {8..15, 24..31}
__device__ __forceinline__ v16bf load_a_frag(const bf16_t* base, int stride) {
  const int lane = threadIdx.x & 31;
  const int row  = lane & 15;
  const int k0   = (lane < 16) ? 0 : 8;
  const bf16_t* p = base + (size_t)row * stride + k0;
  AFragU f;
  f.h[0] = *(const v8bf*)(p);
  f.h[1] = *(const v8bf*)(p + 16);
  return f.v;
}

// B fragment: 32x16 bf16 (KxN), source column-contiguous [n][k], stride in elements.
// lanes 0-15 -> N=lane, K = {0..7, 16..23}; lanes 16-31 -> N=lane-16, K = {8..15, 24..31}
__device__ __forceinline__ v16bf load_b_frag(const bf16_t* base, int stride) {
  const int lane = threadIdx.x & 31;
  const int col  = lane & 15;
  const int k0   = (lane < 16) ? 0 : 8;
  const bf16_t* p = base + (size_t)col * stride + k0;
  AFragU f;
  f.h[0] = *(const v8bf*)(p);
  f.h[1] = *(const v8bf*)(p + 16);
  return f.v;
}

__device__ __forceinline__ v8f wmma_bf16(v16bf a, v16bf b, v8f c) {
  return __builtin_amdgcn_wmma_f32_16x16x32_bf16(false, a, false, b, (short)0, c,
                                                 false, false);
}

// ---------------- kernel 0: fp32 -> bf16 convert ----------------
__global__ void cvt_f32_bf16(const float* __restrict__ src, bf16_t* __restrict__ dst, int n) {
  int i = blockIdx.x * blockDim.x + threadIdx.x;
  if (i < n) dst[i] = f2bf(src[i]);
}

// ---------------- kernel 1: QKV projection GEMM -----------------
// C[4096,3072] = ybf x Wqkv^T + bias, scatter to q (scaled), k, vT.
// Block = 256 thr (8 waves x 16 rows), N-tile 64. W tile staged async to LDS.
__global__ void qkv_gemm_kernel(const bf16_t* __restrict__ ybf,
                                const bf16_t* __restrict__ wbf,
                                const float*  __restrict__ bias,
                                bf16_t* __restrict__ qbf,
                                bf16_t* __restrict__ kbf,
                                bf16_t* __restrict__ vT) {
  __shared__ bf16_t wtile[2][64 * 32];   // [buf][col][k] : 2 x 4KB
  const int tid  = threadIdx.x;
  const int lane = tid & 31;
  const int wave = tid >> 5;
  const int m0 = blockIdx.x * 128 + wave * 16;
  const int n0 = blockIdx.y * 64;

  // per-thread staging chunk: 256 threads x 16B == 4KB tile
  const int srow  = tid >> 2;          // 0..63 (W row = output col)
  const int spart = (tid & 3) * 8;     // bf16 offset within row
  const bf16_t* wsrc = wbf + (size_t)(n0 + srow) * D_MODEL + spart;
  const int sdst = srow * 32 + spart;

  v8f acc[4];
#pragma unroll
  for (int t = 0; t < 4; ++t) acc[t] = {};

  const bf16_t* arow = ybf + (size_t)m0 * D_MODEL;

  async_b128(&wtile[0][sdst], wsrc);                 // prologue stage
  for (int kb = 0; kb < D_MODEL; kb += 32) {
    const int cur = (kb >> 5) & 1;
    if (kb + 32 < D_MODEL) {
      async_b128(&wtile[cur ^ 1][sdst], wsrc + kb + 32);
      ASYNC_WAIT(1);
    } else {
      ASYNC_WAIT(0);
    }
    __syncthreads();
    __builtin_prefetch(arow + kb + 512, 0, 1);
    v16bf a = load_a_frag(arow + kb, D_MODEL);
#pragma unroll
    for (int t = 0; t < 4; ++t) {
      v16bf b = load_b_frag(&wtile[cur][t * 16 * 32], 32);
      acc[t] = wmma_bf16(a, b, acc[t]);
    }
    __syncthreads();   // all reads done before buffer reuse
  }

  const int rbase = (lane >> 4) * 8;
  const int col   = lane & 15;
#pragma unroll
  for (int t = 0; t < 4; ++t) {
    const int e  = n0 + t * 16 + col;
    const float bv = bias[e];
    const int h = e / 192;
    const int j = e % 192;
#pragma unroll
    for (int r = 0; r < 8; ++r) {
      const int M = m0 + rbase + r;
      const int b = M >> 11;
      const int s = M & (S_LEN - 1);
      const int bh = b * HEADS + h;
      const float v = acc[t][r] + bv;
      if (j < 64) {
        qbf[((size_t)bh * S_LEN + s) * HEAD_DIM + j] = f2bf(v * 0.125f);   // 1/sqrt(64)
      } else if (j < 128) {
        kbf[((size_t)bh * S_LEN + s) * HEAD_DIM + (j - 64)] = f2bf(v);
      } else {
        vT[((size_t)bh * HEAD_DIM + (j - 128)) * S_LEN + s] = f2bf(v);
      }
    }
  }
}

// ---------------- kernel 2: flash attention ---------------------
// grid.x = B*H, grid.y = S/64; block = 128 (4 waves x 16 queries).
// K/V tiles shared by all waves, staged async to LDS, double-buffered.
__global__ void attn_kernel(const bf16_t* __restrict__ qbf,
                            const bf16_t* __restrict__ kbf,
                            const bf16_t* __restrict__ vT,
                            bf16_t* __restrict__ valbf) {
  __shared__ bf16_t ktile[2][32 * 64];   // [buf][key][d]   2 x 4KB
  __shared__ bf16_t vtile[2][64 * 32];   // [buf][d][key]   2 x 4KB
  __shared__ bf16_t pshare[4][16 * 32];  // per-wave P staging, 4KB

  const int tid  = threadIdx.x;
  const int lane = tid & 31;
  const int wave = tid >> 5;
  const int bh = blockIdx.x;
  const int b  = bh >> 4;
  const int h  = bh & 15;
  const int q0 = blockIdx.y * 64 + wave * 16;

  const bf16_t* qbase = qbf + ((size_t)bh * S_LEN + q0) * HEAD_DIM;
  const bf16_t* kbase = kbf + (size_t)bh * S_LEN * HEAD_DIM;
  const bf16_t* vbase = vT + (size_t)bh * HEAD_DIM * S_LEN;

  // staging chunks: 128 threads, 2 chunks each per tile (256 x 16B = 4KB)
  const int kc0 = tid, kc1 = tid + 128;          // K tile chunks (8/row of 64)
  const int krow0 = kc0 >> 3, kp0 = (kc0 & 7) * 8;
  const int krow1 = kc1 >> 3, kp1 = (kc1 & 7) * 8;
  const int vrow0 = kc0 >> 2, vp0 = (kc0 & 3) * 8;   // V tile chunks (4/row of 32)
  const int vrow1 = kc1 >> 2, vp1 = (kc1 & 3) * 8;

  const v16bf qa0 = load_a_frag(qbase, HEAD_DIM);       // d = 0..31
  const v16bf qa1 = load_a_frag(qbase + 32, HEAD_DIM);  // d = 32..63

  v8f o[4];
#pragma unroll
  for (int t = 0; t < 4; ++t) o[t] = {};
  float mrow[8], lrow[8];
#pragma unroll
  for (int r = 0; r < 8; ++r) { mrow[r] = -1e30f; lrow[r] = 0.0f; }

  bf16_t* pw = pshare[wave];
  const int rbase = (lane >> 4) * 8;
  const int col   = lane & 15;

  // prologue: stage kb = 0 into buffer 0 (4 async ops per wave)
  async_b128(&ktile[0][krow0 * 64 + kp0], kbase + (size_t)krow0 * HEAD_DIM + kp0);
  async_b128(&ktile[0][krow1 * 64 + kp1], kbase + (size_t)krow1 * HEAD_DIM + kp1);
  async_b128(&vtile[0][vrow0 * 32 + vp0], vbase + (size_t)vrow0 * S_LEN + vp0);
  async_b128(&vtile[0][vrow1 * 32 + vp1], vbase + (size_t)vrow1 * S_LEN + vp1);

  for (int kb = 0; kb < S_LEN; kb += 32) {
    const int cur = (kb >> 5) & 1;
    if (kb + 32 < S_LEN) {
      const int nxt = cur ^ 1;
      const bf16_t* kn = kbase + (size_t)(kb + 32) * HEAD_DIM;
      const bf16_t* vn = vbase + (kb + 32);
      async_b128(&ktile[nxt][krow0 * 64 + kp0], kn + (size_t)krow0 * HEAD_DIM + kp0);
      async_b128(&ktile[nxt][krow1 * 64 + kp1], kn + (size_t)krow1 * HEAD_DIM + kp1);
      async_b128(&vtile[nxt][vrow0 * 32 + vp0], vn + (size_t)vrow0 * S_LEN + vp0);
      async_b128(&vtile[nxt][vrow1 * 32 + vp1], vn + (size_t)vrow1 * S_LEN + vp1);
      ASYNC_WAIT(4);   // previous stage (4 ops) complete; next stage in flight
    } else {
      ASYNC_WAIT(0);
    }
    __syncthreads();

    // ---- scores S = (Q/sqrt(d)) K^T for keys kb..kb+31 (from LDS) ----
    const bf16_t* kt = ktile[cur];
    v8f s0 = {}, s1 = {};
    s0 = wmma_bf16(qa0, load_b_frag(kt, HEAD_DIM), s0);
    s0 = wmma_bf16(qa1, load_b_frag(kt + 32, HEAD_DIM), s0);
    s1 = wmma_bf16(qa0, load_b_frag(kt + 16 * HEAD_DIM, HEAD_DIM), s1);
    s1 = wmma_bf16(qa1, load_b_frag(kt + 16 * HEAD_DIM + 32, HEAD_DIM), s1);

    // ---- online softmax (row stats per C-fragment register r) ----
#pragma unroll
    for (int r = 0; r < 8; ++r) {
      float vmax = fmaxf(s0[r], s1[r]);
#pragma unroll
      for (int off = 8; off >= 1; off >>= 1)
        vmax = fmaxf(vmax, __shfl_xor(vmax, off, 32));
      const float mnew = fmaxf(mrow[r], vmax);
      const float corr = __expf(mrow[r] - mnew);
      const float p0 = __expf(s0[r] - mnew);
      const float p1 = __expf(s1[r] - mnew);
      float rs = p0 + p1;
#pragma unroll
      for (int off = 8; off >= 1; off >>= 1)
        rs += __shfl_xor(rs, off, 32);
      lrow[r] = lrow[r] * corr + rs;
      mrow[r] = mnew;
      s0[r] = p0; s1[r] = p1;
      o[0][r] *= corr; o[1][r] *= corr; o[2][r] *= corr; o[3][r] *= corr;
    }

    // ---- stage P (C layout) to LDS row-major [16][32], reload as A frag ----
#pragma unroll
    for (int r = 0; r < 8; ++r) {
      pw[(rbase + r) * 32 + col]      = f2bf(s0[r]);
      pw[(rbase + r) * 32 + 16 + col] = f2bf(s1[r]);
    }
    __syncthreads();
    const v16bf pa = load_a_frag(pw, 32);
    __syncthreads();

    // ---- O += P x V (V tile from LDS) ----
    const bf16_t* vt = vtile[cur];
#pragma unroll
    for (int t = 0; t < 4; ++t) {
      v16bf vb = load_b_frag(vt + t * 16 * 32, 32);
      o[t] = wmma_bf16(pa, vb, o[t]);
    }
    __syncthreads();   // all reads of buffers done before they are overwritten
  }

  // ---- normalize and store values as bf16 [B*S, D] ----
#pragma unroll
  for (int t = 0; t < 4; ++t) {
    const int d = h * HEAD_DIM + t * 16 + col;
#pragma unroll
    for (int r = 0; r < 8; ++r) {
      const int s = q0 + rbase + r;
      const float v = o[t][r] / lrow[r];
      valbf[((size_t)(b * S_LEN + s)) * D_MODEL + d] = f2bf(v);
    }
  }
}

// ---------------- kernel 3: output projection GEMM --------------
__global__ void out_gemm_kernel(const bf16_t* __restrict__ valbf,
                                const bf16_t* __restrict__ wbf,
                                const float*  __restrict__ bias,
                                float* __restrict__ out) {
  __shared__ bf16_t wtile[2][64 * 32];
  const int tid  = threadIdx.x;
  const int lane = tid & 31;
  const int wave = tid >> 5;
  const int m0 = blockIdx.x * 128 + wave * 16;
  const int n0 = blockIdx.y * 64;

  const int srow  = tid >> 2;
  const int spart = (tid & 3) * 8;
  const bf16_t* wsrc = wbf + (size_t)(n0 + srow) * D_MODEL + spart;
  const int sdst = srow * 32 + spart;

  v8f acc[4];
#pragma unroll
  for (int t = 0; t < 4; ++t) acc[t] = {};

  const bf16_t* arow = valbf + (size_t)m0 * D_MODEL;

  async_b128(&wtile[0][sdst], wsrc);
  for (int kb = 0; kb < D_MODEL; kb += 32) {
    const int cur = (kb >> 5) & 1;
    if (kb + 32 < D_MODEL) {
      async_b128(&wtile[cur ^ 1][sdst], wsrc + kb + 32);
      ASYNC_WAIT(1);
    } else {
      ASYNC_WAIT(0);
    }
    __syncthreads();
    __builtin_prefetch(arow + kb + 512, 0, 1);
    v16bf a = load_a_frag(arow + kb, D_MODEL);
#pragma unroll
    for (int t = 0; t < 4; ++t) {
      v16bf b = load_b_frag(&wtile[cur][t * 16 * 32], 32);
      acc[t] = wmma_bf16(a, b, acc[t]);
    }
    __syncthreads();
  }

  const int rbase = (lane >> 4) * 8;
  const int col   = lane & 15;
#pragma unroll
  for (int t = 0; t < 4; ++t) {
    const int e = n0 + t * 16 + col;
    const float bv = bias[e];
#pragma unroll
    for (int r = 0; r < 8; ++r) {
      const int M = m0 + rbase + r;
      out[(size_t)M * D_MODEL + e] = acc[t][r] + bv;
    }
  }
}

// ---------------- launcher --------------------------------------
extern "C" void kernel_launch(void* const* d_in, const int* in_sizes, int n_in,
                              void* d_out, int out_size, void* d_ws, size_t ws_size,
                              hipStream_t stream) {
  (void)in_sizes; (void)n_in; (void)out_size; (void)ws_size;
  const float* y    = (const float*)d_in[0];
  const float* Wqkv = (const float*)d_in[1];
  const float* bqkv = (const float*)d_in[2];
  const float* Wout = (const float*)d_in[3];
  const float* bout = (const float*)d_in[4];
  float* out = (float*)d_out;

  char* ws = (char*)d_ws;
  size_t off = 0;
  auto alloc = [&](size_t bytes) -> void* {
    void* p = ws + off;
    off += (bytes + 255) & ~(size_t)255;
    return p;
  };
  bf16_t* ybf    = (bf16_t*)alloc((size_t)M_ROWS * D_MODEL * 2);
  bf16_t* wqkvbf = (bf16_t*)alloc((size_t)QKV_N * D_MODEL * 2);
  bf16_t* woutbf = (bf16_t*)alloc((size_t)D_MODEL * D_MODEL * 2);
  bf16_t* qbf    = (bf16_t*)alloc((size_t)BATCH * HEADS * S_LEN * HEAD_DIM * 2);
  bf16_t* kbf    = (bf16_t*)alloc((size_t)BATCH * HEADS * S_LEN * HEAD_DIM * 2);
  bf16_t* vTbf   = (bf16_t*)alloc((size_t)BATCH * HEADS * HEAD_DIM * S_LEN * 2);
  bf16_t* valbf  = (bf16_t*)alloc((size_t)M_ROWS * D_MODEL * 2);

  const int CT = 256;
  cvt_f32_bf16<<<(M_ROWS * D_MODEL + CT - 1) / CT, CT, 0, stream>>>(y, ybf, M_ROWS * D_MODEL);
  cvt_f32_bf16<<<(QKV_N * D_MODEL + CT - 1) / CT, CT, 0, stream>>>(Wqkv, wqkvbf, QKV_N * D_MODEL);
  cvt_f32_bf16<<<(D_MODEL * D_MODEL + CT - 1) / CT, CT, 0, stream>>>(Wout, woutbf, D_MODEL * D_MODEL);

  qkv_gemm_kernel<<<dim3(M_ROWS / 128, QKV_N / 64), 256, 0, stream>>>(
      ybf, wqkvbf, bqkv, qbf, kbf, vTbf);

  attn_kernel<<<dim3(BATCH * HEADS, S_LEN / 64), 128, 0, stream>>>(
      qbf, kbf, vTbf, valbf);

  out_gemm_kernel<<<dim3(M_ROWS / 128, D_MODEL / 64), 256, 0, stream>>>(
      valbf, woutbf, bout, out);
}
